// Model_22316650070153
// MI455X (gfx1250) — matrix operand order repaired and checked by reading, not compiled, not run on hardware
//
#include <hip/hip_runtime.h>
#include <hip/hip_bf16.h>
#include <math.h>

typedef __attribute__((ext_vector_type(16))) __bf16 v16bf;
typedef __attribute__((ext_vector_type(8)))  __bf16 v8bf;
typedef __attribute__((ext_vector_type(8)))  float  v8f;

#define B_SZ   2048
#define T_SZ   300
#define DIN    101
#define DREC   100
#define DOUT   2
#define KPAD   128      // K padded (101 or 100 -> 128), zeros in pad
#define NPAD   112      // 7 * 16 N tiles
#define NT     7
#define DT_TAU 0.1f

// ---------------------------------------------------------------------------
// Kernel 1: x_proj[B*T,100] = input[B*T,101] @ W_x_ah^T + b_ah
// bf16 WMMA GEMM, result written (f32) into the hstore region of d_out.
// 4 waves / block, each wave owns one 16-row M tile.
// ---------------------------------------------------------------------------
__launch_bounds__(128)
__global__ void xproj_kernel(const float* __restrict__ input,
                             const float* __restrict__ Wx,    // [100][101]
                             const float* __restrict__ bah,   // [100]
                             float* __restrict__ xproj)       // [B*T][100]
{
    __shared__ __align__(16) __bf16 lds_w[NPAD * KPAD];       // B[k][n]=Wx[n][k], stored [n][k]
    __shared__ __align__(16) __bf16 lds_a[4][16 * KPAD];      // per-wave A staging

    const int tid  = threadIdx.x;
    const int wave = tid >> 5;
    const int lane = tid & 31;

    // Cooperative: W_x -> bf16 B-fragment-friendly layout (zero-padded)
    for (int i = tid; i < NPAD * KPAD; i += 128) {
        const int n = i >> 7;
        const int k = i & (KPAD - 1);
        const float v = (n < DREC && k < DIN) ? Wx[n * DIN + k] : 0.0f;
        lds_w[i] = (__bf16)v;
    }

    // Per-wave: stage 16 input rows as bf16, K padded to 128 with zeros
    const long long tile = (long long)blockIdx.x * 4 + wave;
    const long long row0 = tile * 16;
    for (int i = lane; i < 16 * KPAD; i += 32) {
        const int m = i >> 7;
        const int k = i & (KPAD - 1);
        const float v = (k < DIN) ? input[(row0 + m) * DIN + k] : 0.0f;
        lds_a[wave][i] = (__bf16)v;
    }
    __syncthreads();

    const int nl  = lane & 15;
    const int s8  = (lane >> 4) << 3;   // A-frag K sub-offset (0 / 8)
    const int hi8 = (lane >> 4) << 3;   // C-layout row offset (0 / 8)
    const int s16 = (lane >> 4) << 4;   // unused; B frag uses s8 (0 / 8) like A

    v8f acc[NT] = {};

#pragma unroll
    for (int kb = 0; kb < 4; ++kb) {
        // A fragment: 16-bit A 16x32 layout -> two 16B LDS loads per lane
        const __bf16* ap = &lds_a[wave][nl * KPAD + kb * 32 + s8];
        const v8bf alo = *(const v8bf*)ap;
        const v8bf ahi = *(const v8bf*)(ap + 16);
        v16bf A;
#pragma unroll
        for (int i = 0; i < 8; ++i) { A[i] = alo[i]; A[i + 8] = ahi[i]; }

#pragma unroll
        for (int c = 0; c < NT; ++c) {
            const __bf16* bp = &lds_w[(c * 16 + nl) * KPAD + kb * 32 + s8];
            const v8bf blo = *(const v8bf*)bp;
            const v8bf bhi = *(const v8bf*)(bp + 16);
            v16bf Bf;
#pragma unroll
            for (int i = 0; i < 8; ++i) { Bf[i] = blo[i]; Bf[i + 8] = bhi[i]; }
            acc[c] = __builtin_amdgcn_wmma_f32_16x16x32_bf16(
                false, A, false, Bf, (short)0, acc[c], false, false);
        }
    }

    // Bias add + f32 store (C layout: reg r -> row r(+8), lane -> n)
#pragma unroll
    for (int c = 0; c < NT; ++c) {
        const int n = c * 16 + nl;
        if (n < DREC) {
            const float bv = bah[n];
#pragma unroll
            for (int r = 0; r < 8; ++r) {
                const long long row = row0 + r + hi8;
                xproj[row * DREC + n] = acc[c][r] + bv;
            }
        }
    }
}

// ---------------------------------------------------------------------------
// Kernel 2: sequential scan. One wave (32 threads) per 16 batch rows.
//   ah = 0.9*ah + 0.1*(h @ W_h^T + xproj_t);  h = retanh(ah) + noise_t
// h staged as bf16 in LDS (A layout); W_h as bf16 in LDS (B layout).
// xproj is read from hstore and overwritten in place with h.
// ---------------------------------------------------------------------------
__launch_bounds__(32)
__global__ void scan_kernel(const float* __restrict__ noise,   // [B,T,100]
                            const float* __restrict__ Wh,      // [100][100]
                            const float* __restrict__ Wy,      // [2][100]
                            const float* __restrict__ ah0,     // [100]
                            float* __restrict__ out,           // [B,T,2]
                            float* __restrict__ hstore)        // [B,T,100] (holds xproj)
{
    __shared__ __align__(16) __bf16 lds_wh[NPAD * KPAD];
    __shared__ __align__(16) __bf16 hstage[16 * KPAD];

    const int lane = threadIdx.x;
    const int b0   = blockIdx.x * 16;

    // W_h -> bf16 B-layout (zero-padded N and K)
    for (int i = lane; i < NPAD * KPAD; i += 32) {
        const int n = i >> 7;
        const int k = i & (KPAD - 1);
        const float v = (n < DREC && k < DREC) ? Wh[n * DREC + k] : 0.0f;
        lds_wh[i] = (__bf16)v;
    }
    // h_init = retanh(ah0), K pad zeroed (stays zero forever)
    for (int i = lane; i < 16 * KPAD; i += 32) {
        const int k = i & (KPAD - 1);
        const float v = (k < DREC) ? fmaxf(tanhf(ah0[k]), 0.0f) : 0.0f;
        hstage[i] = (__bf16)v;
    }

    const int nl  = lane & 15;
    const int hi8 = (lane >> 4) << 3;
    const int s16 = (lane >> 4) << 4;

    float wy0[NT], wy1[NT];
    v8f ah[NT];
#pragma unroll
    for (int c = 0; c < NT; ++c) {
        const int n = c * 16 + nl;
        const bool valid = n < DREC;
        wy0[c] = valid ? Wy[n] : 0.0f;
        wy1[c] = valid ? Wy[DREC + n] : 0.0f;
        const float a0 = valid ? ah0[n] : 0.0f;
#pragma unroll
        for (int r = 0; r < 8; ++r) ah[c][r] = a0;
    }
    __syncthreads();

    for (int t = 0; t < T_SZ; ++t) {
        // ---- acc = h @ W_h^T via bf16 WMMA (28 per step) ----
        v8f acc[NT] = {};
#pragma unroll
        for (int kb = 0; kb < 4; ++kb) {
            const __bf16* ap = &hstage[nl * KPAD + kb * 32 + hi8];
            const v8bf alo = *(const v8bf*)ap;
            const v8bf ahi = *(const v8bf*)(ap + 16);
            v16bf A;
#pragma unroll
            for (int i = 0; i < 8; ++i) { A[i] = alo[i]; A[i + 8] = ahi[i]; }
#pragma unroll
            for (int c = 0; c < NT; ++c) {
                const __bf16* bp = &lds_wh[(c * 16 + nl) * KPAD + kb * 32 + hi8];
                const v8bf blo = *(const v8bf*)bp;
                const v8bf bhi = *(const v8bf*)(bp + 16);
                v16bf Bf;
#pragma unroll
                for (int i = 0; i < 8; ++i) { Bf[i] = blo[i]; Bf[i + 8] = bhi[i]; }
                acc[c] = __builtin_amdgcn_wmma_f32_16x16x32_bf16(
                    false, A, false, Bf, (short)0, acc[c], false, false);
            }
        }

        // ---- elementwise update; xproj read then overwritten with h ----
        float hval[NT][8];
#pragma unroll
        for (int c = 0; c < NT; ++c) {
            const int  n     = c * 16 + nl;
            const bool valid = n < DREC;
#pragma unroll
            for (int r = 0; r < 8; ++r) {
                const long long idx =
                    ((long long)(b0 + r + hi8) * T_SZ + t) * DREC + n;
                const float xp = valid ? hstore[idx] : 0.0f;
                const float nz = valid ? noise[idx]  : 0.0f;
                const float a  = 0.9f * ah[c][r] + DT_TAU * (acc[c][r] + xp);
                ah[c][r] = a;
                const float h = fmaxf(tanhf(a), 0.0f) + nz;
                hval[c][r] = h;
                if (valid) hstore[idx] = h;   // in-place replace xproj with h
            }
        }

        // ---- output projection (D_out=2) with 16-lane xor reduction ----
#pragma unroll
        for (int r = 0; r < 8; ++r) {
            float s0 = 0.0f, s1 = 0.0f;
#pragma unroll
            for (int c = 0; c < NT; ++c) {
                s0 += hval[c][r] * wy0[c];
                s1 += hval[c][r] * wy1[c];
            }
#pragma unroll
            for (int m = 8; m >= 1; m >>= 1) {
                s0 += __shfl_xor(s0, m, 16);
                s1 += __shfl_xor(s1, m, 16);
            }
            if (nl == 0) {
                const long long ob =
                    ((long long)(b0 + r + hi8) * T_SZ + t) * DOUT;
                out[ob]     = s0;
                out[ob + 1] = s1;
            }
        }

        // ---- stage h (bf16) for next step; pad columns stay zero ----
#pragma unroll
        for (int c = 0; c < NT; ++c) {
            const int n = c * 16 + nl;
            if (n < DREC) {
#pragma unroll
                for (int r = 0; r < 8; ++r)
                    hstage[(r + hi8) * KPAD + n] = (__bf16)hval[c][r];
            }
        }
        __syncthreads();   // single-wave WG: orders LDS store -> next-step load
    }
}

// ---------------------------------------------------------------------------
extern "C" void kernel_launch(void* const* d_in, const int* in_sizes, int n_in,
                              void* d_out, int out_size, void* d_ws, size_t ws_size,
                              hipStream_t stream) {
    (void)in_sizes; (void)n_in; (void)d_ws; (void)ws_size; (void)out_size;

    const float* input = (const float*)d_in[0];   // [B,T,101]
    const float* noise = (const float*)d_in[1];   // [B,T,100]
    const float* Wx    = (const float*)d_in[2];   // [100,101]
    const float* bah   = (const float*)d_in[3];   // [100]
    const float* Wh    = (const float*)d_in[4];   // [100,100]
    const float* Wy    = (const float*)d_in[5];   // [2,100]
    const float* ah0   = (const float*)d_in[6];   // [100]

    float* out    = (float*)d_out;                          // [B,T,2]
    float* hstore = out + (long long)B_SZ * T_SZ * DOUT;    // [B,T,100]

    // Phase 1: x_proj -> hstore region (f32), 16-row tiles, 4 waves/block
    const int mtiles = (B_SZ * T_SZ) / 16;   // 38400
    xproj_kernel<<<dim3(mtiles / 4), dim3(128), 0, stream>>>(input, Wx, bah, hstore);

    // Phase 2: sequential scan, one wave per 16 batch rows
    scan_kernel<<<dim3(B_SZ / 16), dim3(32), 0, stream>>>(noise, Wh, Wy, ah0, out, hstore);
}
